// LongConv_80788334837960
// MI455X (gfx1250) — matrix operand order repaired and checked by reading, not compiled, not run on hardware
//
#include <hip/hip_runtime.h>

// ---------------------------------------------------------------------------
// Problem constants (from the reference)
// ---------------------------------------------------------------------------
#define SEQ     8192
#define INCH    64
#define OUTCH   64
#define CH      4
#define BATCH   4
#define NK      7
#define KDIM    128
#define NTAP    770          // 6*128 full segments + 2 taps of segment 6
#define QDIM    (SEQ + 1)    // prefix length per (b,h)

#if __has_builtin(__builtin_amdgcn_global_load_async_to_lds_b128) && \
    __has_builtin(__builtin_amdgcn_s_wait_asynccnt)
#define LC_ASYNC 1
#else
#define LC_ASYNC 0
#endif

typedef __attribute__((ext_vector_type(16))) __bf16        v16bf;
typedef __attribute__((ext_vector_type(8)))  float         v8f;
typedef __attribute__((ext_vector_type(4)))  float         v4f;
typedef __attribute__((ext_vector_type(4)))  unsigned int  v4u;
typedef __attribute__((ext_vector_type(8)))  unsigned int  v8u;

// async-DMA pointer types: builtin wants int __vector(4) pointers in AS1/AS3
typedef int v4ig __attribute__((vector_size(16)));
typedef __attribute__((address_space(1))) v4ig* gvp_t;
typedef __attribute__((address_space(3))) v4ig* lvp_t;

union BFrag { v8u u; v16bf v; };
union AFrag { struct { v4u lo; v4u hh; } p; v16bf v; };

// One packed-pair bf16 convert: cvt(a)->[15:0], cvt(b)->[31:16].
// Mnemonic/operand form verified from the compiler's own gfx1250 output.
__device__ __forceinline__ unsigned int pack_bf16x2(float a, float b) {
  unsigned int u;
  asm("v_cvt_pk_bf16_f32 %0, %1, %2" : "=v"(u) : "v"(a), "v"(b));
  return u;
}

// ---------------------------------------------------------------------------
// 1) norm^2 partials: ||kern||_F^2 of the truncated multi-res kernel.
//    element (i,c,h,j): weight = rep(i,j) * 0.25^i ; rep = 2^i (i<6), i=6: 64 for j<2 else 0
// ---------------------------------------------------------------------------
__global__ __launch_bounds__(256) void lc_norm_part(const float* __restrict__ k,
                                                    float* __restrict__ part) {
  int idx = blockIdx.x * 256 + threadIdx.x;       // 896*256 == 7*4*64*128
  int i = idx >> 15;
  int j = idx & 127;
  float v = k[idx];
  float sc = (i < 6) ? ldexpf(1.0f, -i)           // 2^i * 4^-i = 2^-i
                     : ((j < 2) ? (1.0f / 64.0f) : 0.0f);
  float c = v * v * sc;
  __shared__ float red[256];
  red[threadIdx.x] = c;
  __syncthreads();
  for (int o = 128; o > 0; o >>= 1) {
    if (threadIdx.x < o) red[threadIdx.x] += red[threadIdx.x + o];
    __syncthreads();
  }
  if (threadIdx.x == 0) part[blockIdx.x] = red[0];
}

__global__ __launch_bounds__(256) void lc_norm_reduce(const float* __restrict__ part,
                                                      float* __restrict__ out) {
  float s = 0.0f;
  for (int u = threadIdx.x; u < 896; u += 256) s += part[u];  // fixed order: deterministic
  __shared__ float red[256];
  red[threadIdx.x] = s;
  __syncthreads();
  for (int o = 128; o > 0; o >>= 1) {
    if (threadIdx.x < o) red[threadIdx.x] += red[threadIdx.x + o];
    __syncthreads();
  }
  if (threadIdx.x == 0) out[0] = red[0];
}

// ---------------------------------------------------------------------------
// 2) prefix sums, stored transposed: P[b][q][h], q=0..8192, P[b][0][h]=0
// ---------------------------------------------------------------------------
__global__ __launch_bounds__(256) void lc_prefix(const float* __restrict__ x,
                                                 float* __restrict__ P) {
  int bh = blockIdx.x;                 // b*64 + h
  int b  = bh >> 6;
  int h  = bh & 63;
  int t  = threadIdx.x;
  const float* xp = x + (size_t)bh * SEQ + t * 32;
  float loc[32];
  float s = 0.0f;
#pragma unroll
  for (int u = 0; u < 32; ++u) { s += xp[u]; loc[u] = s; }
  __shared__ float ps[256];
  ps[t] = s;
  __syncthreads();
  for (int off = 1; off < 256; off <<= 1) {
    float add = (t >= off) ? ps[t - off] : 0.0f;
    __syncthreads();
    ps[t] += add;
    __syncthreads();
  }
  float base = ps[t] - s;              // exclusive prefix of chunk sums
  float* Pb = P + (size_t)b * QDIM * INCH + h;
  if (t == 0) Pb[0] = 0.0f;
#pragma unroll
  for (int u = 0; u < 32; ++u) {
    int q = t * 32 + u + 1;
    Pb[(size_t)q * INCH] = base + loc[u];
  }
}

// ---------------------------------------------------------------------------
// 3) Effective per-tap mixed kernel, bf16, tile layout [tap][o][h]
//    Keff = (2^-i / ||kern||) * sum_c mix_w[o, c*64+h] * kernels[i,c,h,j]
// ---------------------------------------------------------------------------
__global__ __launch_bounds__(256) void lc_keff(const float* __restrict__ kern,
                                               const float* __restrict__ w,
                                               const float* __restrict__ norm2,
                                               unsigned short* __restrict__ Keff) {
  int tap = blockIdx.x;
  int i = (tap < 768) ? (tap >> 7) : 6;
  int j = (tap < 768) ? (tap & 127) : (tap - 768);
  float s = ldexpf(1.0f, -i) / sqrtf(norm2[0]);
  unsigned short* dst = Keff + (size_t)tap * (OUTCH * INCH);
  for (int e = threadIdx.x; e < OUTCH * INCH; e += 256) {
    int o = e >> 6, h = e & 63;
    float a = 0.0f;
#pragma unroll
    for (int c = 0; c < CH; ++c)
      a += w[o * (CH * INCH) + c * INCH + h] *
           kern[(((size_t)i * CH + c) * INCH + h) * KDIM + j];
    dst[e] = (unsigned short)(pack_bf16x2(a * s, 0.0f) & 0xFFFFu);
  }
}

// ---------------------------------------------------------------------------
// 4) Main: per (batch, 128-wide l tile), 8 waves. For each of 770 taps:
//    D[o,l] += Keff_tap[o,h] * (P[q1][h] - P[q0][h])  via v_wmma_f32_16x16x32_bf16
//    q0 of tap j is q1 of tap j+1 within a segment -> carry one row in regs.
//    Keff tiles double-buffered in LDS via async global->LDS DMA (ASYNCcnt).
// ---------------------------------------------------------------------------
__global__ __launch_bounds__(256) void lc_main(const float* __restrict__ P,
                                               const unsigned short* __restrict__ Keff,
                                               const float* __restrict__ bias,
                                               float* __restrict__ out) {
  __shared__ v4u As[2][512];                      // 2 x 64x64 bf16 tile = 16KB
  const int t  = threadIdx.x;
  const int wv = t >> 5;                          // wave id 0..7 -> l sub-block
  const int L  = t & 31;
  const int m  = L & 15;                          // row (A) / col (B,D)
  const int hi = L >> 4;
  const int b  = blockIdx.y;
  const int l  = blockIdx.x * 128 + wv * 16 + m;  // this lane's output column

  const v8f vzero = {0.f, 0.f, 0.f, 0.f, 0.f, 0.f, 0.f, 0.f};
  v8f acc[4];
#pragma unroll
  for (int ob = 0; ob < 4; ++ob) acc[ob] = vzero;

  const float* Pb = P + (size_t)b * QDIM * INCH;

#if LC_ASYNC
  {  // prologue: kick off tap 0's tile into buffer 0 (each 16B chunk per lane)
    gvp_t g  = (gvp_t)(void*)const_cast<unsigned short*>(Keff);
    lvp_t lp = (lvp_t)(void*)&As[0][0];
    __builtin_amdgcn_global_load_async_to_lds_b128(g + t,       lp + t,       0, 0);
    __builtin_amdgcn_global_load_async_to_lds_b128(g + t + 256, lp + t + 256, 0, 0);
  }
#endif

#pragma unroll 1
  for (int i = 0; i < NK; ++i) {
    const int r = 1 << i;
    const int off = (r - 1) << 7;                 // 128*(2^i - 1)
    const int jmax = (i < 6) ? 128 : 2;
    const int tapbase = (i < 6) ? i * 128 : 768;

    int q1 = l - off + 1;
    if (q1 < 0) q1 = 0;
    v4f pc[8];                                    // carried P row (q1), both k-steps
    {
      const float* P1 = Pb + (size_t)q1 * INCH;
#pragma unroll
      for (int ks = 0; ks < 2; ++ks)
#pragma unroll
        for (int u = 0; u < 4; ++u)
          pc[ks * 4 + u] = *(const v4f*)(P1 + ks * 32 + hi * 16 + u * 4);
    }

#pragma unroll 1
    for (int j = 0; j < jmax; ++j) {
      const int tap = tapbase + j;

#if LC_ASYNC
      const int cur = tap & 1;
      __builtin_amdgcn_s_wait_asynccnt(0);        // my chunks of buf[cur] landed
      __syncthreads();                            // whole tile landed; buf[cur^1] free
      if (tap + 1 < NTAP) {                       // prefetch next tile into free buffer
        gvp_t g  = (gvp_t)(void*)const_cast<unsigned short*>(
                       Keff + (size_t)(tap + 1) * (OUTCH * INCH));
        lvp_t lp = (lvp_t)(void*)&As[cur ^ 1][0];
        __builtin_amdgcn_global_load_async_to_lds_b128(g + t,       lp + t,       0, 0);
        __builtin_amdgcn_global_load_async_to_lds_b128(g + t + 256, lp + t + 256, 0, 0);
      }
      const unsigned short* Ash = (const unsigned short*)&As[cur][0];
#else
      __syncthreads();
      const v4u* Kt = (const v4u*)(Keff + (size_t)tap * (OUTCH * INCH));
      As[0][t]       = Kt[t];
      As[0][t + 256] = Kt[t + 256];
      __syncthreads();
      const unsigned short* Ash = (const unsigned short*)&As[0][0];
#endif

      int q0 = q1 - r;
      if (q0 < 0) q0 = 0;
      v4f p0[8];

#pragma unroll
      for (int ks = 0; ks < 2; ++ks) {
        const float* P0r = Pb + (size_t)q0 * INCH + ks * 32 + hi * 16;
#pragma unroll
        for (int u = 0; u < 4; ++u) p0[ks * 4 + u] = *(const v4f*)(P0r + u * 4);

        // B fragment: slot s holds Box(h = ks*32 + hi*16 + s) at column m
        BFrag bf;
#pragma unroll
        for (int v = 0; v < 8; ++v) {
          const int s0 = 2 * v, s1 = 2 * v + 1;
          float b0 = pc[ks * 4 + (s0 >> 2)][s0 & 3] - p0[ks * 4 + (s0 >> 2)][s0 & 3];
          float b1 = pc[ks * 4 + (s1 >> 2)][s1 & 3] - p0[ks * 4 + (s1 >> 2)][s1 & 3];
          bf.u[v] = pack_bf16x2(b0, b1);
        }

        // prefetch ALL four A fragments first (one ds clause), then WMMA chain
        AFrag af[4];
#pragma unroll
        for (int ob = 0; ob < 4; ++ob) {
          const int o = ob * 16 + m;
          af[ob].p.lo = *(const v4u*)(Ash + o * 64 + ks * 32 + hi * 16);
          af[ob].p.hh = *(const v4u*)(Ash + o * 64 + ks * 32 + hi * 16 + 8);
        }
#pragma unroll
        for (int ob = 0; ob < 4; ++ob) {
          acc[ob] = __builtin_amdgcn_wmma_f32_16x16x32_bf16(
              false, af[ob].v, false, bf.v, (short)0, acc[ob], false, false);
        }
      }

#pragma unroll
      for (int u = 0; u < 8; ++u) pc[u] = p0[u];  // q0 row becomes next tap's q1 row
      q1 = q0;
    }
  }

  // store D + bias: VGPR v holds M = v + 8*hi, column = m
#pragma unroll
  for (int ob = 0; ob < 4; ++ob)
#pragma unroll
    for (int v = 0; v < 8; ++v) {
      const int o = ob * 16 + v + 8 * hi;
      out[((size_t)(b * OUTCH + o) << 13) + l] = acc[ob][v] + bias[o];
    }
}

// ---------------------------------------------------------------------------
// launcher
// ---------------------------------------------------------------------------
extern "C" void kernel_launch(void* const* d_in, const int* in_sizes, int n_in,
                              void* d_out, int out_size, void* d_ws, size_t ws_size,
                              hipStream_t stream) {
  const float* x    = (const float*)d_in[0];   // (4,64,8192)
  const float* kern = (const float*)d_in[1];   // (7,4,64,128)
  const float* mw   = (const float*)d_in[2];   // (64,256)
  const float* mb   = (const float*)d_in[3];   // (64,)
  float* out = (float*)d_out;                  // (4,64,8192)

  char* ws = (char*)d_ws;
  float* norm2 = (float*)ws;                                   // 1 float
  float* part  = (float*)(ws + 64);                            // 896 floats
  float* P     = (float*)(ws + 4096);                          // 4*8193*64 f32 = 8,389,632 B
  unsigned short* Keff =
      (unsigned short*)(ws + 4096 + (size_t)BATCH * QDIM * INCH * 4);  // 770*4096 bf16

  lc_norm_part  <<<896, 256, 0, stream>>>(kern, part);
  lc_norm_reduce<<<1,   256, 0, stream>>>(part, norm2);
  lc_prefix     <<<BATCH * INCH, 256, 0, stream>>>(x, P);
  lc_keff       <<<NTAP, 256, 0, stream>>>(kern, mw, norm2, Keff);
  lc_main       <<<dim3(SEQ / 128, BATCH), 256, 0, stream>>>(P, Keff, mb, out);
}
